// SelfAttention2D_37692632990280
// MI455X (gfx1250) — matrix operand order repaired and checked by reading, not compiled, not run on hardware
//
#include <hip/hip_runtime.h>
#include <hip/hip_bf16.h>

typedef _Float16 f16;
typedef __attribute__((ext_vector_type(16))) _Float16 v16h;
typedef __attribute__((ext_vector_type(8)))  _Float16 v8h;
typedef __attribute__((ext_vector_type(8)))  float    v8f;

#define BATCH 8
#define CH    128
#define HW    4096            // 64*64 tokens per batch
#define GRP   8
#define EPSV  1e-5f
#define QKSCALE 0.08838834764831845f  // 1/sqrt(128)
#define KB    64              // key block per staging iteration

// workspace layout (bytes)
#define WS_WH  0ul                                   // 3 * C*C f16  = 98304
#define WS_Q   131072ul                              // [B,N,C] f16  = 8 MB
#define WS_K   (WS_Q  + (size_t)BATCH*HW*CH*2)       // [B,N,C] f16
#define WS_VT  (WS_K  + (size_t)BATCH*HW*CH*2)       // [B,C,N] f16
#define WS_O   (WS_VT + (size_t)BATCH*HW*CH*2)       // [B,C,N] f32  = 16 MB

// ---------------------------------------------------------------------------
// gfx1250 async global->LDS copy (16 bytes per lane), tracked by ASYNCcnt.
// lds = 32-bit LDS byte address (low 32 bits of generic pointer), g = global.
// ---------------------------------------------------------------------------
__device__ __forceinline__ void async_ld16(unsigned lds, const void* g) {
  asm volatile("global_load_async_to_lds_b128 %0, %1, off"
               :: "v"(lds), "v"((unsigned long long)g)
               : "memory");
}
__device__ __forceinline__ void wait_async0() {
#if __has_builtin(__builtin_amdgcn_s_wait_asynccnt)
  __builtin_amdgcn_s_wait_asynccnt(0);
#else
  asm volatile("s_wait_asynccnt 0x0" ::: "memory");
#endif
}

// ---------------------------------------------------------------------------
// Kernel 0: fp32 -> f16 weight conversion (Wq|Wk|Wv concatenated)
// ---------------------------------------------------------------------------
__global__ void wcvt_kernel(const float* __restrict__ Wq,
                            const float* __restrict__ Wk,
                            const float* __restrict__ Wv,
                            f16* __restrict__ dst) {
  int i = blockIdx.x * 256 + threadIdx.x;
  if (i < CH * CH) {
    dst[i]               = (f16)Wq[i];
    dst[CH * CH + i]     = (f16)Wk[i];
    dst[2 * CH * CH + i] = (f16)Wv[i];
  }
}

// ---------------------------------------------------------------------------
// Kernel 1: fused QKV projection (1x1 conv) via f16 WMMA
//   x: [B,C,N] fp32.  Out: Q,K as [B,N,C] f16, V as [B,C,N] f16.
//   Block = 256 threads (8 waves), tile = 64 pixels x 128 channels.
// ---------------------------------------------------------------------------
__global__ void __launch_bounds__(256) qkv_proj_kernel(
    const float* __restrict__ x, const f16* __restrict__ Wh,
    const float* __restrict__ bq, const float* __restrict__ bk,
    const float* __restrict__ bv,
    f16* __restrict__ Qh, f16* __restrict__ Kh, f16* __restrict__ Vth) {
  __shared__ f16 Xs[64][CH + 8];  // X^T tile, [pixel][channel], padded
  const int tid = threadIdx.x;
  const int bb  = blockIdx.y;
  const int n0  = blockIdx.x * 64;
  const float* xb = x + (size_t)bb * CH * HW;

  // stage X^T tile into LDS as f16 (global reads coalesced along n)
  for (int i = 0; i < 32; ++i) {
    int idx = tid + i * 256;
    int c = idx >> 6, nn = idx & 63;
    Xs[nn][c] = (f16)xb[(size_t)c * HW + n0 + nn];
  }
  __syncthreads();

  const int lane = tid & 31;
  const int l15  = lane & 15;
  const int hi   = lane >> 4;
  const int o0   = (tid >> 5) * 16;  // each wave owns one 16-wide output tile

  for (int nt = 0; nt < 4; ++nt) {
    // A fragments: X^T rows n0+16*nt .. +15, four K=32 steps
    v16h af[4];
    const int row = nt * 16 + l15;
    #pragma unroll
    for (int cs = 0; cs < 4; ++cs) {
      const f16* p = &Xs[row][cs * 32 + hi * 16];
      v8h lo = *(const v8h*)p;
      v8h h8 = *(const v8h*)(p + 8);
      #pragma unroll
      for (int j = 0; j < 8; ++j) { af[cs][j] = lo[j]; af[cs][8 + j] = h8[j]; }
    }
    #pragma unroll
    for (int m = 0; m < 3; ++m) {
      const f16* W = Wh + (size_t)m * CH * CH;
      const float* bias = (m == 0) ? bq : (m == 1) ? bk : bv;
      float bval = bias[o0 + l15];
      v8f acc;
      #pragma unroll
      for (int r = 0; r < 8; ++r) acc[r] = bval;
      #pragma unroll
      for (int cs = 0; cs < 4; ++cs) {
        // B fragment: 16 contiguous halves of row (o0+l15) of W
        const f16* wp = W + (size_t)(o0 + l15) * CH + cs * 32 + hi * 16;
        v8h b0 = *(const v8h*)wp;
        v8h b1 = *(const v8h*)(wp + 8);
        v16h bf;
        #pragma unroll
        for (int j = 0; j < 8; ++j) { bf[j] = b0[j]; bf[8 + j] = b1[j]; }
        acc = __builtin_amdgcn_wmma_f32_16x16x32_f16(false, af[cs], false, bf,
                                                     (short)0, acc, false, false);
      }
      if (m < 2) {  // Q,K stored [N,C]
        f16* out = (m == 0 ? Qh : Kh) + (size_t)bb * HW * CH;
        #pragma unroll
        for (int r = 0; r < 8; ++r) {
          int n = n0 + nt * 16 + r + hi * 8;
          out[(size_t)n * CH + o0 + l15] = (f16)acc[r];
        }
      } else {      // V stored transposed [C,N] -> contiguous 8-half store/lane
        f16* out = Vth + (size_t)bb * CH * HW;
        v8h vv;
        #pragma unroll
        for (int r = 0; r < 8; ++r) vv[r] = (f16)acc[r];
        *(v8h*)&out[(size_t)(o0 + l15) * HW + n0 + nt * 16 + hi * 8] = vv;
      }
    }
  }
}

// ---------------------------------------------------------------------------
// Kernel 2: flash attention.  4 waves/block, each wave owns 16 query rows.
//   64-key blocks DMA'd into LDS with global_load_async_to_lds_b128;
//   online softmax; O -> [B,C,N] fp32.
// ---------------------------------------------------------------------------
__global__ void __launch_bounds__(128) attn_kernel(
    const f16* __restrict__ Qh, const f16* __restrict__ Kh,
    const f16* __restrict__ Vth, float* __restrict__ O) {
  __shared__ f16 Ks[KB][CH];       // [key][channel]          16 KB
  __shared__ f16 Vs[CH][KB];       // [channel][key] (from V^T) 16 KB
  __shared__ f16 Ps[4][16][KB];    // per-wave P relayout buf    8 KB
  const int tid  = threadIdx.x;
  const int bb   = blockIdx.y;
  const int wave = tid >> 5;
  const int lane = tid & 31;
  const int l15  = lane & 15;
  const int hi   = lane >> 4;
  const int q0   = blockIdx.x * 64 + wave * 16;
  const f16* Qb = Qh  + (size_t)bb * HW * CH;
  const f16* Kb = Kh  + (size_t)bb * HW * CH;
  const f16* Vb = Vth + (size_t)bb * CH * HW;
  const unsigned ksBase = (unsigned)(unsigned long long)&Ks[0][0];
  const unsigned vsBase = (unsigned)(unsigned long long)&Vs[0][0];

  // Q fragments held in registers for the whole sweep
  v16h qa[4];
  {
    const f16* qp = Qb + (size_t)(q0 + l15) * CH;
    #pragma unroll
    for (int cs = 0; cs < 4; ++cs) {
      v8h lo = *(const v8h*)(qp + cs * 32 + hi * 16);
      v8h h8 = *(const v8h*)(qp + cs * 32 + 8 + hi * 16);
      #pragma unroll
      for (int j = 0; j < 8; ++j) { qa[cs][j] = lo[j]; qa[cs][8 + j] = h8[j]; }
    }
  }

  float mi[8], li[8];
  v8f accs[8];
  #pragma unroll
  for (int r = 0; r < 8; ++r) { mi[r] = -1e30f; li[r] = 0.f; }
  #pragma unroll
  for (int t = 0; t < 8; ++t)
    #pragma unroll
    for (int r = 0; r < 8; ++r) accs[t][r] = 0.f;

  for (int m0 = 0; m0 < HW; m0 += KB) {
    __syncthreads();   // everyone done reading previous K/V block
    // async DMA stage: 1024 16-byte chunks each for Ks and Vs, 8 per thread
    #pragma unroll
    for (int i = 0; i < 8; ++i) {
      int ci = tid + i * 128;
      async_ld16(ksBase + ci * 16,
                 Kb + (size_t)(m0 + (ci >> 4)) * CH + (ci & 15) * 8);
      async_ld16(vsBase + ci * 16,
                 Vb + (size_t)(ci >> 3) * HW + m0 + (ci & 7) * 8);
    }
    wait_async0();
    __syncthreads();

    // S = Q K^T for 16x64 tile (four 16x16 accumulators)
    v8f s[4];
    #pragma unroll
    for (int t = 0; t < 4; ++t)
      #pragma unroll
      for (int r = 0; r < 8; ++r) s[t][r] = 0.f;
    #pragma unroll
    for (int cs = 0; cs < 4; ++cs) {
      #pragma unroll
      for (int t = 0; t < 4; ++t) {
        const f16* kp = &Ks[t * 16 + l15][cs * 32 + hi * 16];
        v8h b0 = *(const v8h*)kp, b1 = *(const v8h*)(kp + 8);
        v16h bf;
        #pragma unroll
        for (int j = 0; j < 8; ++j) { bf[j] = b0[j]; bf[8 + j] = b1[j]; }
        s[t] = __builtin_amdgcn_wmma_f32_16x16x32_f16(false, qa[cs], false, bf,
                                                      (short)0, s[t], false, false);
      }
    }

    // online softmax (rows spread over lanes 0-15 / 16-31 -> width-16 shfl)
    float alpha[8];
    #pragma unroll
    for (int r = 0; r < 8; ++r) {
      float sv0 = s[0][r] * QKSCALE, sv1 = s[1][r] * QKSCALE;
      float sv2 = s[2][r] * QKSCALE, sv3 = s[3][r] * QKSCALE;
      float v = fmaxf(fmaxf(sv0, sv1), fmaxf(sv2, sv3));
      #pragma unroll
      for (int off = 8; off > 0; off >>= 1) v = fmaxf(v, __shfl_xor(v, off, 16));
      float mn = fmaxf(mi[r], v);
      float al = __expf(mi[r] - mn);
      float p0 = __expf(sv0 - mn), p1 = __expf(sv1 - mn);
      float p2 = __expf(sv2 - mn), p3 = __expf(sv3 - mn);
      float ps = (p0 + p1) + (p2 + p3);
      #pragma unroll
      for (int off = 8; off > 0; off >>= 1) ps += __shfl_xor(ps, off, 16);
      li[r] = li[r] * al + ps;
      mi[r] = mn;
      alpha[r] = al;
      Ps[wave][r + hi * 8][l15]      = (f16)p0;   // D-layout -> LDS
      Ps[wave][r + hi * 8][l15 + 16] = (f16)p1;
      Ps[wave][r + hi * 8][l15 + 32] = (f16)p2;
      Ps[wave][r + hi * 8][l15 + 48] = (f16)p3;
    }
    #pragma unroll
    for (int t = 0; t < 8; ++t)
      #pragma unroll
      for (int r = 0; r < 8; ++r) accs[t][r] *= alpha[r];

    // reload P in A-fragment layout (per-wave region, DS ops in-order)
    v16h pa0, pa1;
    {
      const f16* pp = &Ps[wave][l15][0];
      v8h a0 = *(const v8h*)(pp + hi * 8);
      v8h a1 = *(const v8h*)(pp + 16 + hi * 8);
      v8h a2 = *(const v8h*)(pp + 32 + hi * 8);
      v8h a3 = *(const v8h*)(pp + 48 + hi * 8);
      #pragma unroll
      for (int j = 0; j < 8; ++j) {
        pa0[j] = a0[j]; pa0[8 + j] = a1[j];
        pa1[j] = a2[j]; pa1[8 + j] = a3[j];
      }
    }
    // O += P V  (8 channel tiles of 16, K=64 as two K=32 steps)
    #pragma unroll
    for (int t = 0; t < 8; ++t) {
      const f16* vp = &Vs[t * 16 + l15][0];
      v8h b0 = *(const v8h*)(vp + hi * 8);
      v8h b1 = *(const v8h*)(vp + 16 + hi * 8);
      v8h b2 = *(const v8h*)(vp + 32 + hi * 8);
      v8h b3 = *(const v8h*)(vp + 48 + hi * 8);
      v16h bf0, bf1;
      #pragma unroll
      for (int j = 0; j < 8; ++j) {
        bf0[j] = b0[j]; bf0[8 + j] = b1[j];
        bf1[j] = b2[j]; bf1[8 + j] = b3[j];
      }
      accs[t] = __builtin_amdgcn_wmma_f32_16x16x32_f16(false, pa0, false, bf0,
                                                       (short)0, accs[t], false, false);
      accs[t] = __builtin_amdgcn_wmma_f32_16x16x32_f16(false, pa1, false, bf1,
                                                       (short)0, accs[t], false, false);
    }
  }

  // epilogue: normalize, store O as [C,N] fp32 (contiguous 8 floats per lane)
  float* Ob = O + (size_t)bb * CH * HW;
  #pragma unroll
  for (int t = 0; t < 8; ++t) {
    float* dst = &Ob[(size_t)(t * 16 + l15) * HW + q0 + hi * 8];
    #pragma unroll
    for (int r = 0; r < 8; ++r) dst[r] = accs[t][r] / li[r];
  }
}

// ---------------------------------------------------------------------------
// Kernel 3: out = GroupNorm(gamma*O + x).  One block per (group, batch).
// ---------------------------------------------------------------------------
__global__ void __launch_bounds__(256) gn_kernel(
    const float* __restrict__ x, const float* __restrict__ O,
    const float* __restrict__ gamma, const float* __restrict__ gw,
    const float* __restrict__ gb, float* __restrict__ out) {
  __shared__ float rs[256], rs2[256];
  const int tid = threadIdx.x;
  const int g = blockIdx.x, bb = blockIdx.y;
  const int CPG = CH / GRP;                 // 16 channels per group
  const int NEL = CPG * HW;                 // 65536 elements
  const size_t base = (size_t)bb * CH * HW + (size_t)g * CPG * HW;
  const float gm = gamma[0];

  float s = 0.f, s2 = 0.f;
  for (int i = tid * 4; i < NEL; i += 1024) {
    float4 xv = *(const float4*)(x + base + i);
    float4 ov = *(const float4*)(O + base + i);
    float y0 = gm * ov.x + xv.x, y1 = gm * ov.y + xv.y;
    float y2 = gm * ov.z + xv.z, y3 = gm * ov.w + xv.w;
    s  += y0 + y1 + y2 + y3;
    s2 += y0 * y0 + y1 * y1 + y2 * y2 + y3 * y3;
  }
  rs[tid] = s; rs2[tid] = s2;
  __syncthreads();
  for (int off = 128; off > 0; off >>= 1) {
    if (tid < off) { rs[tid] += rs[tid + off]; rs2[tid] += rs2[tid + off]; }
    __syncthreads();
  }
  const float mean = rs[0] / (float)NEL;
  const float var  = rs2[0] / (float)NEL - mean * mean;
  const float rinv = rsqrtf(var + EPSV);

  for (int i = tid * 4; i < NEL; i += 1024) {
    int c = g * CPG + (i >> 12);            // HW = 4096 = 2^12
    float wv = gw[c], bv = gb[c];
    float4 xv = *(const float4*)(x + base + i);
    float4 ov = *(const float4*)(O + base + i);
    float4 r;
    r.x = (gm * ov.x + xv.x - mean) * rinv * wv + bv;
    r.y = (gm * ov.y + xv.y - mean) * rinv * wv + bv;
    r.z = (gm * ov.z + xv.z - mean) * rinv * wv + bv;
    r.w = (gm * ov.w + xv.w - mean) * rinv * wv + bv;
    *(float4*)(out + base + i) = r;
  }
}

// ---------------------------------------------------------------------------
extern "C" void kernel_launch(void* const* d_in, const int* in_sizes, int n_in,
                              void* d_out, int out_size, void* d_ws, size_t ws_size,
                              hipStream_t stream) {
  const float* x     = (const float*)d_in[0];
  const float* Wq    = (const float*)d_in[1];
  const float* bq    = (const float*)d_in[2];
  const float* Wk    = (const float*)d_in[3];
  const float* bk    = (const float*)d_in[4];
  const float* Wv    = (const float*)d_in[5];
  const float* bv    = (const float*)d_in[6];
  const float* gamma = (const float*)d_in[7];
  const float* gw    = (const float*)d_in[8];
  const float* gb    = (const float*)d_in[9];
  float* out = (float*)d_out;
  char*  ws  = (char*)d_ws;

  f16*   Wh  = (f16*)(ws + WS_WH);
  f16*   Qh  = (f16*)(ws + WS_Q);
  f16*   Kh  = (f16*)(ws + WS_K);
  f16*   Vth = (f16*)(ws + WS_VT);
  float* O   = (float*)(ws + WS_O);

  wcvt_kernel<<<(CH * CH + 255) / 256, 256, 0, stream>>>(Wq, Wk, Wv, Wh);
  qkv_proj_kernel<<<dim3(HW / 64, BATCH), 256, 0, stream>>>(x, Wh, bq, bk, bv,
                                                            Qh, Kh, Vth);
  attn_kernel<<<dim3(HW / 64, BATCH), 128, 0, stream>>>(Qh, Kh, Vth, O);
  gn_kernel<<<dim3(GRP, BATCH), 256, 0, stream>>>(x, O, gamma, gw, gb, out);
}
